// SNU_Regression_60739427500525
// MI455X (gfx1250) — matrix operand order repaired and checked by reading, not compiled, not run on hardware
//
#include <hip/hip_runtime.h>
#include <hip/hip_bf16.h>

// SNU recurrent conv net for MI455X (gfx1250, wave32).
// - State arrays (~134MB) stay L2-resident (192MB L2); x streamed with NT loads.
// - conv2 = implicit GEMM on v_wmma_f32_16x16x32_bf16 with a custom K-permutation
//   k' = dy*12 + dx*4 + i over a channel-interleaved LDS tile, so every A-fragment
//   4-element chunk is a single 8-byte ds_load_b64.
// - maxpool2 + FC dot fused into the WMMA kernel (block is pool-complete), so y2
//   is never re-read from global; per-block partial dots are reduced
//   deterministically and summed in fixed order by the final scan.

#define L_TAU 0.8f
#define BSZ   128
#define TSZ   20

typedef __attribute__((ext_vector_type(16))) __bf16 v16bf;
typedef __attribute__((ext_vector_type(4)))  __bf16 v4bf;
typedef __attribute__((ext_vector_type(8)))  float  v8f;

__device__ __forceinline__ float elu_f(float x)  { return x > 0.f ? x : (__expf(x) - 1.f); }
__device__ __forceinline__ float sigm_f(float x) { return 1.f / (1.f + __expf(-x)); }

// ---------------------------------------------------------------- zero states
__global__ void snu_zero(float* __restrict__ p, long n) {
    long i = (long)blockIdx.x * blockDim.x + threadIdx.x;
    long stride = (long)gridDim.x * blockDim.x;
    for (; i < n; i += stride) p[i] = 0.f;
}

// ------------------------------------------------- layer 1: conv1+SNU+pool
// One thread per (b, pooled pixel): computes 4 out-channels over a 2x2 block,
// updates s1/y1 state in-place, writes channel-interleaved bf16 p1[B,64,64,4]
// as a single 8-byte store.
__global__ void snu_layer1(const float* __restrict__ x,   // [B,2,T,128,128]
                           const float* __restrict__ W1,  // [4,2,3,3]
                           const float* __restrict__ b1,  // [4]
                           float* __restrict__ s1, float* __restrict__ y1, // [B,4,128,128]
                           v4bf* __restrict__ p1,         // [B,64,64,(4ch)]
                           int t)
{
    __shared__ float w1s[72];
    __shared__ float b1s[4];
    if (threadIdx.x < 72) w1s[threadIdx.x] = W1[threadIdx.x];
    if (threadIdx.x < 4)  b1s[threadIdx.x] = b1[threadIdx.x];
    __syncthreads();

    int tid = blockIdx.x * blockDim.x + threadIdx.x;   // B*64*64 threads
    int b  = tid >> 12;
    int pp = tid & 4095;
    int py = pp >> 6, px = pp & 63;
    int y0 = py << 1, x0 = px << 1;

    // 4x4 input patches (2 channels), zero-padded at image borders.
    float xp[2][4][4];
    const long xbase = (long)b * (2l * TSZ * 16384l) + (long)t * 16384l;
#pragma unroll
    for (int i = 0; i < 2; i++) {
        const float* xc = x + xbase + (long)i * (TSZ * 16384l);
#pragma unroll
        for (int r = 0; r < 4; r++) {
            int yy = y0 - 1 + r;
#pragma unroll
            for (int c = 0; c < 4; c++) {
                int xx = x0 - 1 + c;
                float v = 0.f;
                if ((unsigned)yy < 128u && (unsigned)xx < 128u)
                    v = __builtin_nontemporal_load(xc + yy * 128 + xx); // stream x, keep L2 for state
                xp[i][r][c] = v;
            }
        }
    }

    const long sbase = (long)b * (4l * 16384l);
    v4bf pv;
#pragma unroll
    for (int o = 0; o < 4; o++) {
        float pool = -1e30f;
#pragma unroll
        for (int ry = 0; ry < 2; ry++)
#pragma unroll
        for (int rx = 0; rx < 2; rx++) {
            float acc = 0.f;
#pragma unroll
            for (int i = 0; i < 2; i++)
#pragma unroll
            for (int dy = 0; dy < 3; dy++)
#pragma unroll
            for (int dx = 0; dx < 3; dx++)
                acc = fmaf(xp[i][ry + dy][rx + dx], w1s[o * 18 + i * 9 + dy * 3 + dx], acc);

            long si = sbase + (long)o * 16384l + (long)(y0 + ry) * 128 + (x0 + rx);
            float sOld = s1[si], yOld = y1[si];
            float sNew = elu_f(acc + L_TAU * sOld * (1.f - yOld));
            float yNew = sigm_f(sNew + b1s[o]);
            s1[si] = sNew; y1[si] = yNew;
            pool = fmaxf(pool, yNew);
        }
        pv[o] = (__bf16)pool;
    }
    p1[(long)b * 4096l + py * 64 + px] = pv;   // one 8B store, channel-interleaved
}

// -------------------------------- layer 2: WMMA conv2 + SNU + pool2 + FC dot
// Implicit GEMM: D[16 pixels x 16 channels] = patches[16 x K=36] * W2'[36 x 16]
// via two v_wmma_f32_16x16x32_bf16 (K padded 36 -> 64), K permuted as
// k' = dy*12 + dx*4 + i so A-fragment chunks are contiguous ds_load_b64.
// Block: 256 threads (8 waves), grid (B, 4): block handles rows [16*by,16*by+16),
// which is pool-complete, so maxpool + Wout dot happen in-block (no y2 re-read).
__global__ void snu_layer2_wmma(const v4bf* __restrict__ p1,    // [B,64,64,(4ch)]
                                const float* __restrict__ W2,   // [16,4,3,3]
                                const float* __restrict__ b2,   // [16]
                                float* __restrict__ s2, float* __restrict__ y2, // [B,16,64,64]
                                const float* __restrict__ Wout, // [16384,1]
                                float* __restrict__ dot,        // [B*4, T] partials
                                int t)
{
    // Channel-interleaved, zero-padded tile: 18 rows x 66 cols x 4ch bf16 (v4bf).
    __shared__ v4bf  tile4[18 * 66];
    __shared__ float poolx[16 * 16 * 32];  // [ch][local row][X] x-direction maxes
    __shared__ float red[256];
    const int b  = blockIdx.x;
    const int r0 = blockIdx.y << 4;
    const int tid = threadIdx.x;

    v4bf z4; z4[0] = z4[1] = z4[2] = z4[3] = (__bf16)0.f;

    const v4bf* pb = p1 + (long)b * 4096l;
    for (int p = tid; p < 18 * 66; p += 256) {       // b64 global -> b64 LDS copy
        int lr = p / 66, cc = p % 66;
        int gr = r0 - 1 + lr, gc = cc - 1;
        v4bf v = z4;
        if ((unsigned)gr < 64u && (unsigned)gc < 64u) v = pb[gr * 64 + gc];
        tile4[p] = v;
    }
    __syncthreads();

    const int lane = tid & 31, wave = tid >> 5;
    const int n = lane & 15;           // output channel column (B/C fragment N)

    // B fragments with the same K-permutation: k' -> i = k'&3, dx = (k'%12)>>2, dy = k'/12.
    // Layout (matches A): lanes 0-15 hold K=e for e<8, K=e+8 for e>=8; lanes 16-31 add 8 (N=lane%16).
    v16bf bf0, bf1;
#pragma unroll
    for (int e = 0; e < 16; e++) {
        int k0 = (e & 7) + ((e >> 3) << 4) + (lane < 16 ? 0 : 8);
        int k1 = k0 + 32;
        __bf16 w0 = (__bf16)0.f, w1 = (__bf16)0.f;
        if (k0 < 36) {
            int i = k0 & 3, dx = (k0 % 12) >> 2, dy = k0 / 12;
            w0 = (__bf16)W2[n * 36 + i * 9 + dy * 3 + dx];
        }
        if (k1 < 36) {
            int i = k1 & 3, dx = (k1 % 12) >> 2, dy = k1 / 12;
            w1 = (__bf16)W2[n * 36 + i * 9 + dy * 3 + dx];
        }
        bf0[e] = w0; bf1[e] = w1;
    }
    const float bias = b2[n];
    const long  sbase = (long)b * (16l * 4096l);

    // 64 groups of 16 pixels (16 rows x 4 col-groups), 8 per wave.
    for (int g = (blockIdx.y << 6) + wave; g < (blockIdx.y << 6) + 64; g += 8) {
        const int row  = g >> 2;
        const int col0 = (g & 3) << 4;
        const int M    = lane & 15;        // pixel index within group (A fragment M)
        const int colb = col0 + M;

        // 8-byte chunk loader: k'-run [kp, kp+3] is contiguous in the interleaved
        // tile at v4bf index (row-r0+dy)*66 + colb + pos/4, pos = kp%12 in {0,4,8}.
        auto ldchunk = [&](int kp) -> v4bf {
            if (kp < 36) {
                int dy = kp / 12, pos = kp % 12;
                return tile4[(row - r0 + dy) * 66 + colb + (pos >> 2)];
            }
            return z4;
        };

        // A fragments (16-bit A 16x32 layout): lane<16 covers K kb+0..7, kb+16..23
        // with kb=0; lanes 16-31 use kb=8. frag1 adds +32 (mostly zero pad).
        const int kb = (lane < 16) ? 0 : 8;
        v4bf c0 = ldchunk(kb + 0),  c1 = ldchunk(kb + 4);
        v4bf c2 = ldchunk(kb + 16), c3 = ldchunk(kb + 20);
        v4bf d0 = ldchunk(kb + 32), d1 = ldchunk(kb + 36);
        v4bf d2 = ldchunk(kb + 48), d3 = ldchunk(kb + 52);
        v16bf a0, a1;
#pragma unroll
        for (int e = 0; e < 4; e++) {
            a0[e] = c0[e]; a0[4 + e] = c1[e]; a0[8 + e] = c2[e]; a0[12 + e] = c3[e];
            a1[e] = d0[e]; a1[4 + e] = d1[e]; a1[8 + e] = d2[e]; a1[12 + e] = d3[e];
        }

        v8f c = {0.f, 0.f, 0.f, 0.f, 0.f, 0.f, 0.f, 0.f};
        c = __builtin_amdgcn_wmma_f32_16x16x32_bf16(false, a0, false, bf0, (short)0, c, false, false);
        c = __builtin_amdgcn_wmma_f32_16x16x32_bf16(false, a1, false, bf1, (short)0, c, false, false);

        // Fused SNU update on the accumulator. C layout: VGPR r -> M = r (+8 for
        // lanes 16-31), N = lane%16. Each lane owns 8 adjacent-column elements.
        float yv[8];
#pragma unroll
        for (int r = 0; r < 8; r++) {
            int  pix = r + (lane < 16 ? 0 : 8);
            long si  = sbase + (long)n * 4096l + row * 64 + (col0 + pix);
            float sOld = s2[si], yOld = y2[si];
            float sNew = elu_f(c[r] + L_TAU * sOld * (1.f - yOld));
            float yNew = sigm_f(sNew + bias);
            s2[si] = sNew; y2[si] = yNew;
            yv[r] = yNew;
        }
        // x-direction pool: lane's 8 columns are adjacent -> 4 register maxes.
        const int Xb = (col0 >> 1) + (lane < 16 ? 0 : 4);
#pragma unroll
        for (int j = 0; j < 4; j++)
            poolx[(n * 16 + (row - r0)) * 32 + Xb + j] = fmaxf(yv[2 * j], yv[2 * j + 1]);
    }
    __syncthreads();

    // y-direction pool + Wout dot over this block's 16ch x 8Y x 32X pooled tile.
    float acc = 0.f;
    const int Yg = r0 >> 1;                       // global pooled-row base = by*8
#pragma unroll
    for (int it = 0; it < 16; it++) {
        int idx = it * 256 + tid;                 // 0..4095: ch*256 + yloc*32 + X
        int ch   = idx >> 8;
        int rem  = idx & 255;
        int yloc = rem >> 5, X = rem & 31;
        float v = fmaxf(poolx[(ch * 16 + 2 * yloc) * 32 + X],
                        poolx[(ch * 16 + 2 * yloc + 1) * 32 + X]);
        acc = fmaf(v, Wout[ch * 1024 + (Yg + yloc) * 32 + X], acc);
    }
    red[tid] = acc;
    __syncthreads();
#pragma unroll
    for (int s = 128; s > 0; s >>= 1) {
        if (tid < s) red[tid] += red[tid + s];
        __syncthreads();
    }
    if (tid == 0) dot[((b << 2) | blockIdx.y) * TSZ + t] = red[0];
}

// ------------------------------------------------- s3 linear scan over time
// Sums the 4 row-strip partials in fixed order (deterministic), then scans.
__global__ void snu_scan(const float* __restrict__ dot, float* __restrict__ out) {
    int b = threadIdx.x;
    if (b < BSZ) {
        float s3 = 0.f;
#pragma unroll
        for (int t = 0; t < TSZ; t++) {
            float d = ((dot[(b * 4 + 0) * TSZ + t]  + dot[(b * 4 + 1) * TSZ + t])
                     + dot[(b * 4 + 2) * TSZ + t]) + dot[(b * 4 + 3) * TSZ + t];
            s3 = d + L_TAU * s3;
            out[b * TSZ + t] = s3;
        }
    }
}

// ------------------------------------------------------------------ launcher
extern "C" void kernel_launch(void* const* d_in, const int* in_sizes, int n_in,
                              void* d_out, int out_size, void* d_ws, size_t ws_size,
                              hipStream_t stream) {
    const float* x    = (const float*)d_in[0];
    // d_in[1] = y labels, unused by forward math
    const float* W1   = (const float*)d_in[2];
    const float* b1   = (const float*)d_in[3];
    const float* W2   = (const float*)d_in[4];
    const float* b2   = (const float*)d_in[5];
    const float* Wout = (const float*)d_in[6];
    float* out = (float*)d_out;

    // Workspace layout (~132 MiB):
    const long N1 = (long)BSZ * 4 * 128 * 128;  // 8,388,608 (s1/y1)
    const long N2 = (long)BSZ * 16 * 64 * 64;   // 8,388,608 (s2/y2)
    char* ws = (char*)d_ws;
    float* s1  = (float*)ws;  ws += N1 * sizeof(float);
    float* y1  = (float*)ws;  ws += N1 * sizeof(float);
    float* s2  = (float*)ws;  ws += N2 * sizeof(float);
    float* y2  = (float*)ws;  ws += N2 * sizeof(float);
    float* dot = (float*)ws;  ws += (long)BSZ * 4 * TSZ * sizeof(float);
    v4bf*  p1  = (v4bf*)ws;   // BSZ*64*64 v4bf (channel-interleaved), 4 MiB

    // Zero recurrent state + dot every call (harness only poisons ws once).
    const long nz = N1 * 2 + N2 * 2 + (long)BSZ * 4 * TSZ;
    snu_zero<<<2048, 256, 0, stream>>>((float*)d_ws, nz);

    const int l1_blocks = (BSZ * 64 * 64) / 256;  // 2048
    for (int t = 0; t < TSZ; t++) {
        snu_layer1<<<l1_blocks, 256, 0, stream>>>(x, W1, b1, s1, y1, p1, t);
        snu_layer2_wmma<<<dim3(BSZ, 4), 256, 0, stream>>>(p1, W2, b2, s2, y2,
                                                          Wout, dot, t);
    }
    snu_scan<<<1, 128, 0, stream>>>(dot, out);
}
